// CA3_RecurrentMatrix_40192303956586
// MI455X (gfx1250) — compile-verified, matrix operand order checked by reading, not run on hardware
//
#include <hip/hip_runtime.h>
#include <math.h>

// ---------------------------------------------------------------------------
// CA3 recurrent attractor: 5 steps of cosine-sim softmax attention.
// B=256 queries, C=100000 memory rows, D=1024.
//
// MI455X reasoning:
//  * ~524 GFLOP total, memory bank = 205 MB in f16. Fused one-pass-per-step
//    flash-style kernel: ~44us HBM time vs ~105us dense-f16 WMMA time ->
//    compute bound; f16 (v_wmma_f32_16x16x32_f16) is the right precision
//    (all operands are unit-normalized, |sim|<=1, so f16's 10 mantissa bits
//    beat bf16 and fp8 here while still using the matrix pipes).
//  * sims bounded in [-1,1] -> exp() is overflow-safe -> no online max;
//    split-C softmax partials combine with plain sums (deterministic).
//  * This revision: batched B-fragment preloads (one load clause + staggered
//    loadcnt waits per 4 WMMAs), merged stage/GEMM1 phase (one fewer barrier
//    per chunk), and global_prefetch of chunk ch+1 under chunk ch's compute.
// ---------------------------------------------------------------------------

typedef __attribute__((ext_vector_type(16))) _Float16 v16h;
typedef __attribute__((ext_vector_type(8)))  float    v8f;

#define B_Q     256
#define C_MEM   100000
#define D_DIM   1024
#define STEPS   5
#define EPSN    1e-8f
#define NSPLIT  25            // C split across workgroups: 100000/25 = 4000
#define QTILES  16            // 256 queries / 16
#define CN      32            // memory rows per chunk (K of GEMM2 WMMA)
#define NCHUNK  125           // 4000 / 32
#define SMT     40            // padded LDS stride (halves) for transposed chunk
#define SP      40            // padded LDS stride for P' tile

// Load one 16-bit A/B WMMA fragment (K-major operand). Per ISA 7.12.2:
// lane half selects K-base (+0 / +8), elements 0..7 = K..K+7, 8..15 = K+16..K+23.
// Caller passes p = row_base + kbase + khi; we fetch p[0..7] and p[16..23].
__device__ __forceinline__ v16h frag_ld(const _Float16* p) {
    v16h r;
    reinterpret_cast<float4*>(&r)[0] = *reinterpret_cast<const float4*>(p);
    reinterpret_cast<float4*>(&r)[1] = *reinterpret_cast<const float4*>(p + 16);
    return r;
}

__device__ __forceinline__ v8f wmma_f16(v16h a, v16h b, v8f c) {
    return __builtin_amdgcn_wmma_f32_16x16x32_f16(false, a, false, b,
                                                  (short)0, c, false, false);
}

// --------------------------------------------------------------------------
// Kernel 1: per-row normalize memory bank -> f16 Mhat, f32 norms. One WG/row.
// --------------------------------------------------------------------------
__global__ __launch_bounds__(256) void prep_mem_kernel(
        const float* __restrict__ M, _Float16* __restrict__ mhat,
        float* __restrict__ mnorm) {
    const int c = blockIdx.x;
    const int tid = threadIdx.x;
    const float* row = M + (size_t)c * D_DIM;
    float4 x = *reinterpret_cast<const float4*>(row + tid * 4);
    float ss = x.x * x.x + x.y * x.y + x.z * x.z + x.w * x.w;
#pragma unroll
    for (int m = 16; m >= 1; m >>= 1) ss += __shfl_xor(ss, m, 32);
    __shared__ float red[8];
    if ((tid & 31) == 0) red[tid >> 5] = ss;
    __syncthreads();
    float tot = 0.f;
#pragma unroll
    for (int i = 0; i < 8; ++i) tot += red[i];
    const float nrm = fmaxf(sqrtf(tot), EPSN);
    const float rs  = 1.0f / nrm;
    union { _Float16 h[4]; float2 f2; } u;
    u.h[0] = (_Float16)(x.x * rs); u.h[1] = (_Float16)(x.y * rs);
    u.h[2] = (_Float16)(x.z * rs); u.h[3] = (_Float16)(x.w * rs);
    *reinterpret_cast<float2*>(mhat + (size_t)c * D_DIM + tid * 4) = u.f2;
    if (tid == 0) mnorm[c] = nrm;
}

// --------------------------------------------------------------------------
// Kernel 2: normalize current queries -> f16 Qhat. One WG/row.
// --------------------------------------------------------------------------
__global__ __launch_bounds__(256) void norm_q_kernel(
        const float* __restrict__ cur, _Float16* __restrict__ qhat) {
    const int b = blockIdx.x;
    const int tid = threadIdx.x;
    const float* row = cur + (size_t)b * D_DIM;
    float4 x = *reinterpret_cast<const float4*>(row + tid * 4);
    float ss = x.x * x.x + x.y * x.y + x.z * x.z + x.w * x.w;
#pragma unroll
    for (int m = 16; m >= 1; m >>= 1) ss += __shfl_xor(ss, m, 32);
    __shared__ float red[8];
    if ((tid & 31) == 0) red[tid >> 5] = ss;
    __syncthreads();
    float tot = 0.f;
#pragma unroll
    for (int i = 0; i < 8; ++i) tot += red[i];
    const float rs = 1.0f / fmaxf(sqrtf(tot), EPSN);
    union { _Float16 h[4]; float2 f2; } u;
    u.h[0] = (_Float16)(x.x * rs); u.h[1] = (_Float16)(x.y * rs);
    u.h[2] = (_Float16)(x.z * rs); u.h[3] = (_Float16)(x.w * rs);
    *reinterpret_cast<float2*>(qhat + (size_t)b * D_DIM + tid * 4) = u.f2;
}

// --------------------------------------------------------------------------
// Kernel 3: fused sim->exp->weighted-accumulate over a C-split.
// Grid (NSPLIT, QTILES), 256 threads = 8 waves.
//   Phase A (one barrier region): stage chunk c-major into LDS (for GEMM2's
//          B operand) + GEMM1 (S = Qhat . Mhat^T, K=1024 split across the 8
//          waves, partials to LDS) + prefetch of chunk ch+1.
//   Phase B: reduce K-partials, exp, scale by ||M_c||  (bounded, no max).
//   Phase C: GEMM2 (O += P' . Mhat); each wave owns a 128-wide D slice with
//          accumulators resident in VGPRs across all 125 chunks.
// --------------------------------------------------------------------------
__global__ __launch_bounds__(256) void attn_kernel(
        const _Float16* __restrict__ qhat, const _Float16* __restrict__ mhat,
        const float* __restrict__ mnorm, float* __restrict__ partial,
        float* __restrict__ denp) {
    __shared__ alignas(16) _Float16 mT[D_DIM * SMT];    // 80 KB  (c-major chunk)
    __shared__ float sred[8][16][32];                   // 16 KB  (S K-partials)
    __shared__ alignas(16) _Float16 plds[16 * SP];      // 1.25KB (P' tile)
    __shared__ float mn[CN];

    const int split = blockIdx.x;
    const int qt    = blockIdx.y;
    const int tid   = threadIdx.x;
    const int wv    = tid >> 5;
    const int lane  = tid & 31;
    const int l15   = lane & 15;
    const int khi   = (lane & 16) ? 8 : 0;   // K-base offset of this lane half
    const int qhi   = (lane & 16) ? 8 : 0;   // C/D row offset of this lane half
    const int c_start = split * (C_MEM / NSPLIT);

    const _Float16* qbase = qhat + (size_t)(qt * 16) * D_DIM;
    const int kb0 = wv * 128;                // this wave's K slice for GEMM1

    // Q A-fragments are chunk-invariant: load once.
    v16h afr[4];
#pragma unroll
    for (int kk = 0; kk < 4; ++kk)
        afr[kk] = frag_ld(qbase + (size_t)l15 * D_DIM + kb0 + kk * 32 + khi);

    v8f o[8] = {};            // output accumulators: rows=16 q, cols=wv*128..+127
    float denloc = 0.f;

    for (int ch = 0; ch < NCHUNK; ++ch) {
        const int c0 = c_start + ch * CN;
        __syncthreads();      // previous chunk's GEMM2 done with mT / plds

        // ---- Phase A: stage transposed chunk + GEMM1 + prefetch ch+1 ----
        // stage chunk c-major: mT[d][c] (pairs of c packed as b32 stores)
#pragma unroll
        for (int i = 0; i < 8; ++i) {
            const int w  = i * 256 + tid;         // 2048 work items
            const int cp = (w & 15) * 2;          // even c within chunk
            const int db = (w >> 4) * 8;          // d octet base
            const _Float16* g0 = mhat + (size_t)(c0 + cp) * D_DIM + db;
            float4 r0 = *reinterpret_cast<const float4*>(g0);
            float4 r1 = *reinterpret_cast<const float4*>(g0 + D_DIM);
            const _Float16* h0 = reinterpret_cast<const _Float16*>(&r0);
            const _Float16* h1 = reinterpret_cast<const _Float16*>(&r1);
#pragma unroll
            for (int j = 0; j < 8; ++j) {
                union { _Float16 h[2]; float f; } pu;
                pu.h[0] = h0[j]; pu.h[1] = h1[j];
                *reinterpret_cast<float*>(&mT[(db + j) * SMT + cp]) = pu.f;
            }
        }
        if (tid < CN) mn[tid] = mnorm[c0 + tid];

        // prefetch next chunk's 64 KB while this chunk computes
        if (ch + 1 < NCHUNK) {
            const char* nb = (const char*)(mhat + (size_t)(c0 + CN) * D_DIM);
            __builtin_prefetch(nb + tid * 256, 0, 3);
            __builtin_prefetch(nb + tid * 256 + 128, 0, 3);
        }

        // GEMM1: wave wv covers K slice [kb0, kb0+128); batched B-frag loads
#pragma unroll
        for (int n = 0; n < 2; ++n) {
            const _Float16* bb =
                mhat + (size_t)(c0 + n * 16 + l15) * D_DIM + kb0 + khi;
            v16h bfr[4];
#pragma unroll
            for (int kk = 0; kk < 4; ++kk) bfr[kk] = frag_ld(bb + kk * 32);
            v8f s = {};
#pragma unroll
            for (int kk = 0; kk < 4; ++kk) s = wmma_f16(afr[kk], bfr[kk], s);
            const int col = n * 16 + l15;
#pragma unroll
            for (int r = 0; r < 8; ++r) sred[wv][qhi + r][col] = s[r];
        }
        __syncthreads();

        // ---- Phase B: reduce K-partials, exp, scale by ||M_c|| ----
        {
            const int q  = tid >> 4;
            const int cA = tid & 15;
            const int cB = cA + 16;
            float s0 = 0.f, s1 = 0.f;
#pragma unroll
            for (int w = 0; w < 8; ++w) { s0 += sred[w][q][cA]; s1 += sred[w][q][cB]; }
            const float e0 = __expf(s0);   // |s| <= 1 -> always safe
            const float e1 = __expf(s1);
            plds[q * SP + cA] = (_Float16)(e0 * mn[cA]);
            plds[q * SP + cB] = (_Float16)(e1 * mn[cB]);
            denloc += e0 + e1;
        }
        __syncthreads();

        // ---- Phase C: o[16 x 128-slice] += P'(16x32) . Mhat(32 x slice) ----
        {
            const v16h pfr = frag_ld(&plds[l15 * SP + khi]);
            v16h bfr[4];
#pragma unroll
            for (int t = 0; t < 4; ++t)
                bfr[t] = frag_ld(&mT[(kb0 + t * 16 + l15) * SMT + khi]);
#pragma unroll
            for (int t = 0; t < 4; ++t) o[t] = wmma_f16(pfr, bfr[t], o[t]);
#pragma unroll
            for (int t = 0; t < 4; ++t)
                bfr[t] = frag_ld(&mT[(kb0 + (t + 4) * 16 + l15) * SMT + khi]);
#pragma unroll
            for (int t = 0; t < 4; ++t) o[t + 4] = wmma_f16(pfr, bfr[t], o[t + 4]);
        }
    }

    // ---- epilogue: partial numerators + split denominator ----
    float* pp = partial + ((size_t)split * B_Q + qt * 16) * D_DIM;
#pragma unroll
    for (int t = 0; t < 8; ++t)
#pragma unroll
        for (int r = 0; r < 8; ++r)
            pp[(size_t)(qhi + r) * D_DIM + kb0 + t * 16 + l15] = o[t][r];

#pragma unroll
    for (int m = 8; m >= 1; m >>= 1) denloc += __shfl_xor(denloc, m, 32);
    if (l15 == 0) denp[split * B_Q + qt * 16 + (tid >> 4)] = denloc;
}

// --------------------------------------------------------------------------
// Kernel 4: deterministic split reduction + convex blend.
// --------------------------------------------------------------------------
__global__ __launch_bounds__(256) void blend_kernel(
        const float* __restrict__ partial, const float* __restrict__ denp,
        const float* __restrict__ src, float* __restrict__ dst) {
    const int idx = blockIdx.x * 256 + threadIdx.x;
    const int b = idx >> 10;
    float num = 0.f, den = 0.f;
#pragma unroll
    for (int s = 0; s < NSPLIT; ++s) num += partial[(size_t)s * B_Q * D_DIM + idx];
#pragma unroll
    for (int s = 0; s < NSPLIT; ++s) den += denp[s * B_Q + b];
    dst[idx] = 0.8f * (num / den) + 0.2f * src[idx];
}

// --------------------------------------------------------------------------
extern "C" void kernel_launch(void* const* d_in, const int* in_sizes, int n_in,
                              void* d_out, int out_size, void* d_ws, size_t ws_size,
                              hipStream_t stream) {
    (void)in_sizes; (void)n_in; (void)out_size; (void)ws_size;
    const float* query = (const float*)d_in[0];     // [B, D] f32
    const float* mem   = (const float*)d_in[1];     // [C, D] f32

    char* ws = (char*)d_ws;
    size_t off = 0;
    _Float16* mhat  = (_Float16*)(ws + off); off += (size_t)C_MEM * D_DIM * 2;  // 204.8 MB
    float*    mnorm = (float*)(ws + off);    off += (size_t)C_MEM * 4;          // 0.4 MB
    _Float16* qhat  = (_Float16*)(ws + off); off += (size_t)B_Q * D_DIM * 2;    // 0.5 MB
    float*    cur   = (float*)(ws + off);    off += (size_t)B_Q * D_DIM * 4;    // 1.0 MB
    float*    part  = (float*)(ws + off);    off += (size_t)NSPLIT * B_Q * D_DIM * 4; // 25 MB
    float*    denp  = (float*)(ws + off);    off += (size_t)NSPLIT * B_Q * 4;

    prep_mem_kernel<<<C_MEM, 256, 0, stream>>>(mem, mhat, mnorm);

    for (int step = 0; step < STEPS; ++step) {
        const float* src = (step == 0) ? query : cur;
        float* dst = (step == STEPS - 1) ? (float*)d_out : cur;
        norm_q_kernel<<<B_Q, 256, 0, stream>>>(src, qhat);
        attn_kernel<<<dim3(NSPLIT, QTILES), 256, 0, stream>>>(qhat, mhat, mnorm,
                                                              part, denp);
        blend_kernel<<<(B_Q * D_DIM) / 256, 256, 0, stream>>>(part, denp, src, dst);
    }
}
